// FreqVSSBlock_23991687315887
// MI455X (gfx1250) — compile-verified, matrix operand order checked by reading, not run on hardware
//
#include <hip/hip_runtime.h>
#include <hip/hip_bf16.h>
#include <math.h>

#define HID 96
#define DMOD 192          // 2*HID
#define DIN 384           // d_inner
#define NST 16            // d_state
#define KDIR 4
#define RRANK 12
#define HH 64
#define WW 64
#define WF 33
#define LTOK (HH*WF)      // 2112
#define NPAD 48           // 44 padded to 48 for WMMA
#define NCH 32            // scan chunks
#define CHL 66            // LTOK / NCH
#define TWO_PI 6.28318530717958647692f

typedef __attribute__((ext_vector_type(16))) _Float16 v16h;
typedef __attribute__((ext_vector_type(8)))  float    v8f;

// ---------------------------------------------------------------- utilities
__global__ void k_f32_to_f16(const float* __restrict__ src, _Float16* __restrict__ dst, int n) {
    int i = blockIdx.x * 256 + threadIdx.x;
    if (i < n) dst[i] = (_Float16)src[i];
}

// x_proj_w (K,44,DIN) -> f16 (K,48,DIN), rows 44..47 zero so WMMA N-padding is exact
__global__ void k_pad_xproj(const float* __restrict__ w, _Float16* __restrict__ dst) {
    int i = blockIdx.x * 256 + threadIdx.x;
    if (i >= KDIR * NPAD * DIN) return;
    int d = i % DIN;
    int n = (i / DIN) % NPAD;
    int k = i / (DIN * NPAD);
    dst[i] = (n < RRANK + 2 * NST) ? (_Float16)w[(k * (RRANK + 2 * NST) + n) * DIN + d]
                                   : (_Float16)0.0f;
}

// ---------------------------------------------------------------- forward FFT
__global__ void k_rfft_w(const float* __restrict__ x, float* __restrict__ f1re, float* __restrict__ f1im) {
    int i = blockIdx.x * 256 + threadIdx.x;           // (h, wf, c)
    if (i >= HH * WF * HID) return;
    int c = i % HID, wf = (i / HID) % WF, h = i / (HID * WF);
    float re = 0.f, im = 0.f;
    float step = -TWO_PI * (float)wf / (float)WW;
    for (int w = 0; w < WW; ++w) {
        float s, ct; __sincosf(step * (float)w, &s, &ct);
        float v = x[(h * WW + w) * HID + c];
        re += v * ct; im += v * s;
    }
    f1re[i] = re; f1im[i] = im;
}

__global__ void k_fft_h(const float* __restrict__ f1re, const float* __restrict__ f1im,
                        float* __restrict__ x1, _Float16* __restrict__ x1h) {
    int i = blockIdx.x * 256 + threadIdx.x;           // (hh, wf, c)
    if (i >= HH * WF * HID) return;
    int c = i % HID, wf = (i / HID) % WF, hh = i / (HID * WF);
    float re = 0.f, im = 0.f;
    float step = TWO_PI * (float)hh / (float)HH;
    for (int h = 0; h < HH; ++h) {
        float s, ct; __sincosf(step * (float)h, &s, &ct);  // e^{-i theta}
        float fr = f1re[(h * WF + wf) * HID + c];
        float fi = f1im[(h * WF + wf) * HID + c];
        re += fr * ct + fi * s;
        im += fi * ct - fr * s;
    }
    re *= (1.0f / 64.0f); im *= (1.0f / 64.0f);
    int l = hh * WF + wf;
    x1[(size_t)l * DMOD + 2 * c]     = re;
    x1[(size_t)l * DMOD + 2 * c + 1] = im;
    x1h[(size_t)l * DMOD + 2 * c]     = (_Float16)re;
    x1h[(size_t)l * DMOD + 2 * c + 1] = (_Float16)im;
}

// ---------------------------------------------------------------- WMMA GEMMs
__device__ __forceinline__ void wmma_store_tile(float* __restrict__ C, int ldc,
                                                int m0, int n0, v8f acc) {
    int lane = threadIdx.x & 31;
    int col = lane & 15, half = lane >> 4;
#pragma unroll
    for (int r = 0; r < 8; ++r)
        C[(size_t)(m0 + half * 8 + r) * ldc + n0 + col] = acc[r];
}

// in_proj: xz(L,768) = x1h(L,192) x in_proj_w^T   (W stored (768,192) row-major)
__global__ void k_gemm_inproj(const _Float16* __restrict__ A, const _Float16* __restrict__ W,
                              float* __restrict__ C) {
    int wave = threadIdx.x >> 5;
    int m0 = (blockIdx.x * 4 + wave) * 16, n0 = blockIdx.y * 16;
    int lane = threadIdx.x & 31;
    int row = lane & 15, half = lane >> 4;
    v8f acc = {};
#pragma unroll
    for (int k0 = 0; k0 < DMOD; k0 += 32) {
        v16h a, b;
#pragma unroll
        for (int v = 0; v < 8; ++v) {
            int kk = k0 + ((v & 4) ? 16 : 0) + half * 8 + (v & 3) * 2;
            a[2 * v]     = A[(size_t)(m0 + row) * DMOD + kk];
            a[2 * v + 1] = A[(size_t)(m0 + row) * DMOD + kk + 1];
            b[2 * v]     = W[(size_t)(n0 + row) * DMOD + kk];
            b[2 * v + 1] = W[(size_t)(n0 + row) * DMOD + kk + 1];
        }
        acc = __builtin_amdgcn_wmma_f32_16x16x32_f16(false, a, false, b, (short)0, acc, false, false);
    }
    wmma_store_tile(C, 2 * DIN, m0, n0, acc);
}

// x_dbl: per-direction (L,48) = xs16[k](L,384) x Wp[k]^T  (Wp padded f16 (K,48,384))
__global__ void k_gemm_xdbl(const _Float16* __restrict__ xs16, const _Float16* __restrict__ Wp,
                            float* __restrict__ xdbl) {
    int wave = threadIdx.x >> 5;
    int m0 = (blockIdx.x * 4 + wave) * 16, n0 = blockIdx.y * 16, k = blockIdx.z;
    int lane = threadIdx.x & 31;
    int row = lane & 15, half = lane >> 4;
    const _Float16* A = xs16 + (size_t)k * LTOK * DIN;
    v8f acc = {};
#pragma unroll 2
    for (int k0 = 0; k0 < DIN; k0 += 32) {
        v16h a, b;
#pragma unroll
        for (int v = 0; v < 8; ++v) {
            int kk = k0 + ((v & 4) ? 16 : 0) + half * 8 + (v & 3) * 2;
            a[2 * v]     = A[(size_t)(m0 + row) * DIN + kk];
            a[2 * v + 1] = A[(size_t)(m0 + row) * DIN + kk + 1];
            b[2 * v]     = Wp[(size_t)(k * NPAD + n0 + row) * DIN + kk];
            b[2 * v + 1] = Wp[(size_t)(k * NPAD + n0 + row) * DIN + kk + 1];
        }
        acc = __builtin_amdgcn_wmma_f32_16x16x32_f16(false, a, false, b, (short)0, acc, false, false);
    }
    wmma_store_tile(xdbl + (size_t)k * LTOK * NPAD, NPAD, m0, n0, acc);
}

// out_proj: att(L,192) = yln16(L,384) x out_proj_w^T  (W stored (192,384))
__global__ void k_gemm_outproj(const _Float16* __restrict__ A, const _Float16* __restrict__ W,
                               float* __restrict__ C) {
    int wave = threadIdx.x >> 5;
    int m0 = (blockIdx.x * 4 + wave) * 16, n0 = blockIdx.y * 16;
    int lane = threadIdx.x & 31;
    int row = lane & 15, half = lane >> 4;
    v8f acc = {};
#pragma unroll 2
    for (int k0 = 0; k0 < DIN; k0 += 32) {
        v16h a, b;
#pragma unroll
        for (int v = 0; v < 8; ++v) {
            int kk = k0 + ((v & 4) ? 16 : 0) + half * 8 + (v & 3) * 2;
            a[2 * v]     = A[(size_t)(m0 + row) * DIN + kk];
            a[2 * v + 1] = A[(size_t)(m0 + row) * DIN + kk + 1];
            b[2 * v]     = W[(size_t)(n0 + row) * DIN + kk];
            b[2 * v + 1] = W[(size_t)(n0 + row) * DIN + kk + 1];
        }
        acc = __builtin_amdgcn_wmma_f32_16x16x32_f16(false, a, false, b, (short)0, acc, false, false);
    }
    wmma_store_tile(C, DMOD, m0, n0, acc);
}

// ---------------------------------------------------------------- conv + SiLU
__global__ void k_conv_silu(const float* __restrict__ xz, const float* __restrict__ cw,
                            const float* __restrict__ cb, float* __restrict__ co) {
    int i = blockIdx.x * 256 + threadIdx.x;           // i = d*LTOK + h*WF + w
    if (i >= DIN * LTOK) return;
    int hw = i % LTOK, d = i / LTOK;
    int h = hw / WF, w = hw % WF;
    float acc = cb[d];
#pragma unroll
    for (int dh = -1; dh <= 1; ++dh)
#pragma unroll
        for (int dw = -1; dw <= 1; ++dw) {
            int h2 = h + dh, w2 = w + dw;
            if (h2 >= 0 && h2 < HH && w2 >= 0 && w2 < WF)
                acc += cw[d * 9 + (dh + 1) * 3 + (dw + 1)] * xz[(size_t)(h2 * WF + w2) * (2 * DIN) + d];
        }
    co[i] = acc / (1.0f + __expf(-acc));              // SiLU
}

// ---------------------------------------------------------------- 4-direction gather
__global__ void k_build_xs(const float* __restrict__ co, float* __restrict__ xs32,
                           _Float16* __restrict__ xs16) {
    int i = blockIdx.x * 256 + threadIdx.x;           // ((k*L)+l)*DIN + d
    if (i >= KDIR * LTOK * DIN) return;
    int d = i % DIN;
    int l = (i / DIN) % LTOK;
    int k = i / (DIN * LTOK);
    int lsrc = (k & 2) ? (LTOK - 1 - l) : l;
    int h, w;
    if (k & 1) { w = lsrc / HH; h = lsrc % HH; }      // transposed scan: l = w*H + h
    else       { h = lsrc / WF; w = lsrc % WF; }      // row scan: l = h*Wf + w
    float v = co[(size_t)d * LTOK + h * WF + w];
    xs32[i] = v;
    xs16[i] = (_Float16)v;
}

// ---------------------------------------------------------------- dt proj + softplus
__global__ void k_delta(const float* __restrict__ xdbl, const float* __restrict__ dt_w,
                        const float* __restrict__ dt_b, float* __restrict__ delta) {
    int i = blockIdx.x * 256 + threadIdx.x;           // (k*L+l)*DIN + d
    if (i >= KDIR * LTOK * DIN) return;
    int d = i % DIN;
    int l = (i / DIN) % LTOK;
    int k = i / (DIN * LTOK);
    const float* xr = xdbl + (size_t)(k * LTOK + l) * NPAD;
    const float* wr = dt_w + (size_t)(k * DIN + d) * RRANK;
    float acc = dt_b[k * DIN + d];
#pragma unroll
    for (int r = 0; r < RRANK; ++r) acc += xr[r] * wr[r];
    delta[i] = (acc > 20.0f) ? acc : log1pf(__expf(acc));
}

// ---------------------------------------------------------------- chunked selective scan
// recurrence h_l = a_l*h_{l-1} + b_l, a_l = exp(dt*A), b_l = dt*B_l*u_l (associative)
// pass 1: per (k,d,chunk) compute chunk transition (prod of a, local scan from h=0)
__global__ void k_scan_pass1(const float* __restrict__ xs32, const float* __restrict__ xdbl,
                             const float* __restrict__ delta, const float* __restrict__ A_logs,
                             float* __restrict__ Pc, float* __restrict__ Sc) {
    int t = blockIdx.x * 256 + threadIdx.x;           // t = (k*DIN+d)*NCH + c
    if (t >= KDIR * DIN * NCH) return;
    int c = t % NCH, td = t / NCH;
    int k = td / DIN, d = td % DIN;
    float Aa[NST], P[NST], S[NST];
#pragma unroll
    for (int n = 0; n < NST; ++n) { Aa[n] = -__expf(A_logs[(size_t)td * NST + n]); P[n] = 1.f; S[n] = 0.f; }
    for (int l = c * CHL; l < (c + 1) * CHL; ++l) {
        size_t base = (size_t)(k * LTOK + l);
        float dt = delta[base * DIN + d];
        float u  = xs32[base * DIN + d];
        const float* xr = xdbl + base * NPAD;
        __builtin_prefetch(xr + NPAD, 0, 1);
#pragma unroll
        for (int n = 0; n < NST; ++n) {
            float a = __expf(dt * Aa[n]);
            S[n] = S[n] * a + dt * xr[RRANK + n] * u;
            P[n] *= a;
        }
    }
    size_t o = (size_t)t * NST;
#pragma unroll
    for (int n = 0; n < NST; ++n) { Pc[o + n] = P[n]; Sc[o + n] = S[n]; }
}

// pass 2: tiny prefix over the 32 chunk transitions -> incoming state per chunk
__global__ void k_scan_pass2(const float* __restrict__ Pc, const float* __restrict__ Sc,
                             float* __restrict__ hin) {
    int i = blockIdx.x * 256 + threadIdx.x;           // i = (k*DIN+d)*NST + n
    if (i >= KDIR * DIN * NST) return;
    int n = i % NST, td = i / NST;
    float h = 0.f;
    for (int c = 0; c < NCH; ++c) {
        size_t o = (size_t)(td * NCH + c) * NST + n;
        hin[o] = h;
        h = Pc[o] * h + Sc[o];
    }
}

// pass 3: replay each chunk from its incoming state, emit y
__global__ void k_scan_pass3(const float* __restrict__ xs32, const float* __restrict__ xdbl,
                             const float* __restrict__ delta, const float* __restrict__ A_logs,
                             const float* __restrict__ Ds, const float* __restrict__ hin,
                             float* __restrict__ outy) {
    int t = blockIdx.x * 256 + threadIdx.x;           // t = (k*DIN+d)*NCH + c
    if (t >= KDIR * DIN * NCH) return;
    int c = t % NCH, td = t / NCH;
    int k = td / DIN, d = td % DIN;
    float Aa[NST], hs[NST];
    size_t o = (size_t)t * NST;
#pragma unroll
    for (int n = 0; n < NST; ++n) { Aa[n] = -__expf(A_logs[(size_t)td * NST + n]); hs[n] = hin[o + n]; }
    float Dv = Ds[td];
    for (int l = c * CHL; l < (c + 1) * CHL; ++l) {
        size_t base = (size_t)(k * LTOK + l);
        float dt = delta[base * DIN + d];
        float u  = xs32[base * DIN + d];
        const float* xr = xdbl + base * NPAD;
        __builtin_prefetch(xr + NPAD, 0, 1);
        float y = 0.f;
#pragma unroll
        for (int n = 0; n < NST; ++n) {
            hs[n] = hs[n] * __expf(dt * Aa[n]) + dt * xr[RRANK + n] * u;
            y += hs[n] * xr[RRANK + NST + n];
        }
        outy[base * DIN + d] = y + Dv * u;
    }
}

// ---------------------------------------------------------------- merge dirs + LN + gate
__global__ void k_merge_ln(const float* __restrict__ outy, const float* __restrict__ xz,
                           const float* __restrict__ nw, const float* __restrict__ nb,
                           _Float16* __restrict__ yln16) {
    int l = blockIdx.x;                                // l = h*WF + w
    int d = threadIdx.x;                               // 384 threads
    int h = l / WF, w = l % WF;
    int l1 = w * HH + h;
    float y = outy[((size_t)0 * LTOK + l) * DIN + d]
            + outy[((size_t)2 * LTOK + (LTOK - 1 - l)) * DIN + d]
            + outy[((size_t)1 * LTOK + l1) * DIN + d]
            + outy[((size_t)3 * LTOK + (LTOK - 1 - l1)) * DIN + d];
    __shared__ float red[512];
    __shared__ float mv[2];
    red[d] = y;
    if (d < 128) red[384 + d] = 0.f;
    __syncthreads();
#pragma unroll
    for (int s = 256; s > 0; s >>= 1) {
        if (d < s) red[d] += red[d + s];
        __syncthreads();
    }
    if (d == 0) mv[0] = red[0] / (float)DIN;
    __syncthreads();
    float diff = y - mv[0];
    red[d] = diff * diff;
    if (d < 128) red[384 + d] = 0.f;
    __syncthreads();
#pragma unroll
    for (int s = 256; s > 0; s >>= 1) {
        if (d < s) red[d] += red[d + s];
        __syncthreads();
    }
    if (d == 0) mv[1] = rsqrtf(red[0] / (float)DIN + 1e-5f);
    __syncthreads();
    float ln = diff * mv[1] * nw[d] + nb[d];
    float z = xz[(size_t)l * (2 * DIN) + DIN + d];
    float val = ln * (z / (1.0f + __expf(-z)));
    yln16[(size_t)l * DIN + d] = (_Float16)val;
}

// ---------------------------------------------------------------- inverse FFT
__global__ void k_ifft_h(const float* __restrict__ att, float* __restrict__ gre, float* __restrict__ gim) {
    int i = blockIdx.x * 256 + threadIdx.x;           // (hh, w, c)
    if (i >= HH * WF * HID) return;
    int c = i % HID, w = (i / HID) % WF, hh = i / (HID * WF);
    float re = 0.f, im = 0.f;
    float step = TWO_PI * (float)hh / (float)HH;
    for (int h = 0; h < HH; ++h) {
        float s, ct; __sincosf(step * (float)h, &s, &ct);   // e^{+i theta}
        float ar = att[(size_t)(h * WF + w) * DMOD + 2 * c];
        float ai = att[(size_t)(h * WF + w) * DMOD + 2 * c + 1];
        re += ar * ct - ai * s;
        im += ar * s + ai * ct;
    }
    gre[i] = re; gim[i] = im;
}

__global__ void k_irfft_w(const float* __restrict__ gre, const float* __restrict__ gim,
                          const float* __restrict__ x, float* __restrict__ out) {
    int i = blockIdx.x * 256 + threadIdx.x;           // (hh, ww, c)
    if (i >= HH * WW * HID) return;
    int c = i % HID, ww = (i / HID) % WW, hh = i / (HID * WW);
    const float* gr = gre + (size_t)hh * WF * HID;
    const float* gi = gim + (size_t)hh * WF * HID;
    float acc = gr[0 * HID + c] + ((ww & 1) ? -1.0f : 1.0f) * gr[32 * HID + c];
    float step = TWO_PI * (float)ww / (float)WW;
    for (int w = 1; w < 32; ++w) {
        float s, ct; __sincosf(step * (float)w, &s, &ct);
        acc += 2.0f * (gr[w * HID + c] * ct - gi[w * HID + c] * s);
    }
    out[i] = x[i] + acc * (1.0f / 64.0f);
}

// ---------------------------------------------------------------- launcher
extern "C" void kernel_launch(void* const* d_in, const int* in_sizes, int n_in,
                              void* d_out, int out_size, void* d_ws, size_t ws_size,
                              hipStream_t stream) {
    const float* x         = (const float*)d_in[0];   // (1,64,64,96)
    const float* in_proj_w = (const float*)d_in[1];   // (768,192)
    const float* conv_w    = (const float*)d_in[2];   // (384,1,3,3)
    const float* conv_b    = (const float*)d_in[3];   // (384,)
    const float* x_proj_w  = (const float*)d_in[4];   // (4,44,384)
    const float* dt_w      = (const float*)d_in[5];   // (4,384,12)
    const float* dt_b      = (const float*)d_in[6];   // (4,384)
    const float* A_logs    = (const float*)d_in[7];   // (1536,16)
    const float* Ds        = (const float*)d_in[8];   // (1536,)
    const float* onw       = (const float*)d_in[9];   // (384,)
    const float* onb       = (const float*)d_in[10];  // (384,)
    const float* out_proj_w= (const float*)d_in[11];  // (192,384)
    float* out = (float*)d_out;                       // (1,64,64,96)

    char* ws = (char*)d_ws;
    size_t off = 0;
    auto alloc = [&](size_t bytes) -> void* {
        void* p = ws + off;
        off = (off + bytes + 255) & ~(size_t)255;
        return p;
    };

    float*    x1     = (float*)   alloc((size_t)LTOK * DMOD * 4);
    _Float16* x1h    = (_Float16*)alloc((size_t)LTOK * DMOD * 2);
    _Float16* w_in16 = (_Float16*)alloc((size_t)(2 * DIN) * DMOD * 2);
    _Float16* w_out16= (_Float16*)alloc((size_t)DMOD * DIN * 2);
    _Float16* wxp16  = (_Float16*)alloc((size_t)KDIR * NPAD * DIN * 2);
    float*    xz     = (float*)   alloc((size_t)LTOK * (2 * DIN) * 4);
    float*    convo  = (float*)   alloc((size_t)DIN * LTOK * 4);
    float*    xs32   = (float*)   alloc((size_t)KDIR * LTOK * DIN * 4);
    _Float16* xs16   = (_Float16*)alloc((size_t)KDIR * LTOK * DIN * 2);
    float*    xdbl   = (float*)   alloc((size_t)KDIR * LTOK * NPAD * 4);
    float*    delta  = (float*)   alloc((size_t)KDIR * LTOK * DIN * 4);
    float*    outy   = (float*)   alloc((size_t)KDIR * LTOK * DIN * 4);
    float*    Pc     = (float*)   alloc((size_t)KDIR * DIN * NCH * NST * 4);
    float*    Sc     = (float*)   alloc((size_t)KDIR * DIN * NCH * NST * 4);
    float*    hin    = (float*)   alloc((size_t)KDIR * DIN * NCH * NST * 4);
    _Float16* yln16  = (_Float16*)alloc((size_t)LTOK * DIN * 2);
    float*    att    = (float*)   alloc((size_t)LTOK * DMOD * 4);
    float*    f1re   = (float*)   alloc((size_t)HH * WF * HID * 4);
    float*    f1im   = (float*)   alloc((size_t)HH * WF * HID * 4);
    float*    gre    = f1re;  // disjoint lifetime, reuse
    float*    gim    = f1im;

    auto cdiv = [](int a, int b) { return (a + b - 1) / b; };

    // weight precision conversion
    k_f32_to_f16<<<cdiv(2 * DIN * DMOD, 256), 256, 0, stream>>>(in_proj_w, w_in16, 2 * DIN * DMOD);
    k_f32_to_f16<<<cdiv(DMOD * DIN, 256), 256, 0, stream>>>(out_proj_w, w_out16, DMOD * DIN);
    k_pad_xproj<<<cdiv(KDIR * NPAD * DIN, 256), 256, 0, stream>>>(x_proj_w, wxp16);

    // forward rfft2 (ortho)
    k_rfft_w<<<cdiv(HH * WF * HID, 256), 256, 0, stream>>>(x, f1re, f1im);
    k_fft_h<<<cdiv(HH * WF * HID, 256), 256, 0, stream>>>(f1re, f1im, x1, x1h);

    // in_proj GEMM (WMMA, 4 waves/block each owning a 16-row M tile)
    k_gemm_inproj<<<dim3(LTOK / 64, (2 * DIN) / 16), 128, 0, stream>>>(x1h, w_in16, xz);

    // depthwise conv + SiLU
    k_conv_silu<<<cdiv(DIN * LTOK, 256), 256, 0, stream>>>(xz, conv_w, conv_b, convo);

    // 4-direction cross-scan gather
    k_build_xs<<<cdiv(KDIR * LTOK * DIN, 256), 256, 0, stream>>>(convo, xs32, xs16);

    // x_proj GEMM per direction (WMMA)
    k_gemm_xdbl<<<dim3(LTOK / 64, NPAD / 16, KDIR), 128, 0, stream>>>(xs16, wxp16, xdbl);

    // dt projection + softplus
    k_delta<<<cdiv(KDIR * LTOK * DIN, 256), 256, 0, stream>>>(xdbl, dt_w, dt_b, delta);

    // chunked selective scan: 49k-thread chunk pass, 24k-thread prefix, 49k-thread replay
    k_scan_pass1<<<cdiv(KDIR * DIN * NCH, 256), 256, 0, stream>>>(xs32, xdbl, delta, A_logs, Pc, Sc);
    k_scan_pass2<<<cdiv(KDIR * DIN * NST, 256), 256, 0, stream>>>(Pc, Sc, hin);
    k_scan_pass3<<<cdiv(KDIR * DIN * NCH, 256), 256, 0, stream>>>(xs32, xdbl, delta, A_logs, Ds, hin, outy);

    // merge directions + layernorm + SiLU gate
    k_merge_ln<<<LTOK, DIN, 0, stream>>>(outy, xz, onw, onb, yln16);

    // out_proj GEMM (WMMA)
    k_gemm_outproj<<<dim3(LTOK / 64, DMOD / 16), 128, 0, stream>>>(yln16, w_out16, att);

    // inverse rfft2 (ortho) + residual
    k_ifft_h<<<cdiv(HH * WF * HID, 256), 256, 0, stream>>>(att, gre, gim);
    k_irfft_w<<<cdiv(HH * WW * HID, 256), 256, 0, stream>>>(gre, gim, x, out);
}
